// ResNetBBlock_72662256714583
// MI455X (gfx1250) — matrix operand order repaired and checked by reading, not compiled, not run on hardware
//
#include <hip/hip_runtime.h>
#include <hip/hip_bf16.h>

// ---------------- CDNA5 WMMA plumbing ----------------
typedef __attribute__((ext_vector_type(16))) __bf16 v16bf;
typedef __attribute__((ext_vector_type(2)))  __bf16 v2bf;
typedef __attribute__((ext_vector_type(8)))  float  v8f;
typedef __attribute__((ext_vector_type(2)))  float  v2f;
typedef __attribute__((ext_vector_type(4)))  int    v4i;
typedef unsigned short u16t;
typedef unsigned int   u32t;

#if defined(__has_builtin)
#if __has_builtin(__builtin_amdgcn_wmma_f32_16x16x4_f32)
#define HAS_WMMA_F32F32 1
#endif
#if __has_builtin(__builtin_amdgcn_cvt_pk_bf16_f32)
#define HAS_CVT_PK_BF16 1
#endif
#if __has_builtin(__builtin_amdgcn_global_load_async_to_lds_b128)
#define HAS_ASYNC_LDS 1
#endif
#endif
#ifndef HAS_WMMA_F32F32
#define HAS_WMMA_F32F32 0
#endif
#ifndef HAS_CVT_PK_BF16
#define HAS_CVT_PK_BF16 0
#endif
#ifndef HAS_ASYNC_LDS
#define HAS_ASYNC_LDS 0
#endif

union FragBF { v16bf v; u32t u[8]; };
union FragF  { v8f  v; float f[8]; };

// f32 -> bf16 (RNE). Prefer HW packed converter when clang exposes it.
__device__ __forceinline__ u32t pack_bf16(float lo, float hi) {
#if HAS_CVT_PK_BF16
    union { v2bf v; u32t u; } r;
    r.v = __builtin_amdgcn_cvt_pk_bf16_f32(lo, hi);
    return r.u;
#else
    u32t a = __float_as_uint(lo);
    a = (a + 0x7FFFu + ((a >> 16) & 1u)) >> 16;
    u32t b = __float_as_uint(hi);
    b = (b + 0x7FFFu + ((b >> 16) & 1u)) >> 16;
    return a | (b << 16);
#endif
}
__device__ __forceinline__ u16t f_to_bf16u(float f) {
#if HAS_CVT_PK_BF16
    return (u16t)pack_bf16(f, 0.0f);
#else
    u32t u = __float_as_uint(f);
    return (u16t)((u + 0x7FFFu + ((u >> 16) & 1u)) >> 16);
#endif
}
__device__ __forceinline__ float bf16u_to_f(u16t s) {
    return __uint_as_float(((u32t)s) << 16);
}
__device__ __forceinline__ float lrelu(float x, float s) { return x >= 0.0f ? x : s * x; }

__device__ __forceinline__ v8f wmma_bf16(v16bf a, v16bf b, v8f c) {
    return __builtin_amdgcn_wmma_f32_16x16x32_bf16(false, a, false, b, (short)0, c, false, false);
}
#if HAS_WMMA_F32F32
__device__ __forceinline__ v8f wmma_f32(v2f a, v2f b, v8f c) {
    return __builtin_amdgcn_wmma_f32_16x16x4_f32(false, a, false, b, (short)0, c, false, false);
}
#endif

__device__ __forceinline__ void wait_dscnt0() {
#if defined(__has_builtin) && __has_builtin(__builtin_amdgcn_s_wait_dscnt)
    __builtin_amdgcn_s_wait_dscnt(0);
#else
    asm volatile("s_wait_dscnt 0" ::: "memory");
#endif
}
__device__ __forceinline__ void wait_asynccnt0() {
#if defined(__has_builtin) && __has_builtin(__builtin_amdgcn_s_wait_asynccnt)
    __builtin_amdgcn_s_wait_asynccnt(0);
#else
    asm volatile("s_wait_asynccnt 0" ::: "memory");
#endif
}

#if HAS_ASYNC_LDS
// Async global->LDS B128 copy (GLOBAL_LOAD_ASYNC_TO_LDS_B128, tracked by ASYNCcnt).
// The builtin wants (v4i addrspace(1)*, v4i addrspace(3)*, imm offset, imm cpol).
// AS pointers are produced via integer detours: low 32 bits of a generic LDS
// address are the LDS byte offset (ISA aperture rule), so the truncating cast is exact.
typedef __attribute__((address_space(1))) v4i gv4i_t;
typedef __attribute__((address_space(3))) v4i lv4i_t;
__device__ __forceinline__ void async_ld_lds_b128(const void* gsrc, void* ldst) {
    __builtin_amdgcn_global_load_async_to_lds_b128(
        (gv4i_t*)(uintptr_t)gsrc,
        (lv4i_t*)(unsigned int)(uintptr_t)ldst, 0, 0);
}
#endif

// A-fragment (bf16, 16xK tile) K index of packed pair v (v=0..7) for this lane-half.
__device__ __forceinline__ int a_kpair(int v, int hb) {
    return 16 * hb + 2 * v;
}

#define NPTS   131072          // B*N = 2*65536
#define WAVES  4               // waves per 128-thread block
#define NBLK   512             // grid; 512*4 waves * 4 iters = 8192 tiles

// ---------------- Kernel 1: h = leaky(x @ W_in + b_in, 0.1) -> bf16 ----------------
__global__ __launch_bounds__(128) void k_hidden(const float* __restrict__ x,
                                                const float* __restrict__ Win,
                                                const float* __restrict__ bin,
                                                u16t* __restrict__ h_bf) {
    __shared__ u32t  sWin[32 * 32];      // W_in^T bf16 pairs [N=32][Kpair=32]
    __shared__ float sBin[32];
    __shared__ u32t  sX[WAVES][512];     // per-wave x tile, 16x64 bf16 pairs

    const int tid = threadIdx.x;
    for (int pp = tid; pp < 32 * 32; pp += 128) {
        int N = pp >> 5, q = pp & 31, K = 2 * q;
        sWin[N * 32 + q] = pack_bf16(Win[K * 32 + N], Win[(K + 1) * 32 + N]);
    }
    if (tid < 32) sBin[tid] = bin[tid];
    __syncthreads();

    const int wave = tid >> 5, lane = tid & 31;
    const int hb = lane >> 4, lm = lane & 15;
    const int waveGid = blockIdx.x * WAVES + wave;
    u32t* sx = sX[wave];

    for (int it = 0; it < 4; ++it) {
        const int tile = waveGid + it * (NBLK * WAVES);
        const int P0 = tile * 16;
        for (int q = 0; q < 16; ++q) {
            int pi = lane + 32 * q;
            const float* s = x + (size_t)P0 * 64 + 2 * pi;
            sx[pi] = pack_bf16(s[0], s[1]);
        }
        FragBF a0, a1;
#pragma unroll
        for (int v = 0; v < 8; ++v) {
            int K0 = a_kpair(v, hb);
            a0.u[v] = sx[(lm * 64 + K0) >> 1];
            a1.u[v] = sx[(lm * 64 + K0 + 32) >> 1];
        }
#pragma unroll
        for (int t = 0; t < 2; ++t) {
            int c = t * 16 + lm;
            FragF acc;
            float bv = sBin[c];
#pragma unroll
            for (int j = 0; j < 8; ++j) acc.f[j] = bv;
            FragBF b0, b1;
#pragma unroll
            for (int v = 0; v < 8; ++v) {
                int K = 16 * hb + 2 * v;
                b0.u[v] = sWin[c * 32 + (K >> 1)];
                b1.u[v] = sWin[c * 32 + ((K + 32) >> 1)];
            }
            acc.v = wmma_bf16(a0.v, b0.v, acc.v);
            acc.v = wmma_bf16(a1.v, b1.v, acc.v);
#pragma unroll
            for (int j = 0; j < 8; ++j) {
                int M = j + 8 * hb;
                h_bf[(size_t)(P0 + M) * 32 + c] = f_to_bf16u(lrelu(acc.f[j], 0.1f));
            }
        }
    }
}

// ---------------- Kernel 2: residual + edge MLP + gather-sum + out GEMM ----------------
__global__ __launch_bounds__(128) void k_main(
    const float* __restrict__ x, const float* __restrict__ pos, const int* __restrict__ nidx,
    const float* __restrict__ Wa, const float* __restrict__ ba,
    const float* __restrict__ Wb, const float* __restrict__ bb,
    const float* __restrict__ Wout, const float* __restrict__ bout,
    const float* __restrict__ Wsc, const float* __restrict__ bsc,
    const u16t* __restrict__ h_bf, float* __restrict__ out) {
    __shared__ u32t  sWsc[128 * 32];                 // W_sc^T bf16 pairs  16KB
    __shared__ u32t  sWout[128 * 16];                // W_out^T bf16 pairs  8KB
    __shared__ float sBias[320];                     // ba | bb | b_out | b_sc
    __shared__ u32t  sX[WAVES][512];                 // x tile bf16 pairs (2KB/wave)
    __shared__ float sRel[WAVES][768];               // rel positions (3KB/wave)
    __shared__ int   sIdx[WAVES][256];               // global neighbor ids (1KB/wave)
    __shared__ __align__(16) u16t sConvBF[WAVES][512]; // conv tile bf16 (1KB/wave)
    __shared__ __align__(16) u16t sT[WAVES][512];      // t tile bf16 (1KB/wave)
    __shared__ __align__(16) u16t sHG[WAVES][512];     // gathered h rows bf16 (1KB/wave)

    const int tid = threadIdx.x;
    for (int pp = tid; pp < 128 * 32; pp += 128) {
        int N = pp >> 5, q = pp & 31, K = 2 * q;
        sWsc[N * 32 + q] = pack_bf16(Wsc[K * 128 + N], Wsc[(K + 1) * 128 + N]);
    }
    for (int pp = tid; pp < 128 * 16; pp += 128) {
        int N = pp >> 4, q = pp & 15, K = 2 * q;
        sWout[N * 16 + q] = pack_bf16(Wout[K * 128 + N], Wout[(K + 1) * 128 + N]);
    }
    for (int i = tid; i < 320; i += 128) {
        float v;
        if (i < 32) v = ba[i];
        else if (i < 64) v = bb[i - 32];
        else if (i < 192) v = bout[i - 64];
        else v = bsc[i - 192];
        sBias[i] = v;
    }
    __syncthreads();

    const int wave = tid >> 5, lane = tid & 31;
    const int hb = lane >> 4, lm = lane & 15;
    const int waveGid = blockIdx.x * WAVES + wave;
    u32t*  sx    = sX[wave];
    float* srel  = sRel[wave];
    int*   sidx  = sIdx[wave];
    u16t*  sconv = sConvBF[wave];
    u16t*  st    = sT[wave];
    u16t*  shg   = sHG[wave];

#if HAS_WMMA_F32F32
    v2f bwa[2];                        // Wa B-fragments for V_WMMA_F32_16X16X4_F32
#pragma unroll
    for (int t = 0; t < 2; ++t) {
        int c = t * 16 + lm;
        bwa[t][0] = hb ? Wa[2 * 32 + c] : Wa[0 * 32 + c];
        bwa[t][1] = hb ? 0.0f           : Wa[1 * 32 + c];
    }
#else
    float wac[2][3];
#pragma unroll
    for (int t = 0; t < 2; ++t) {
        int c = t * 16 + lm;
        wac[t][0] = Wa[c]; wac[t][1] = Wa[32 + c]; wac[t][2] = Wa[64 + c];
    }
#endif
    FragBF bwb[2];                     // persistent Wb B-fragments (bf16, K=32)
#pragma unroll
    for (int t = 0; t < 2; ++t) {
        int N = t * 16 + lm;
#pragma unroll
        for (int v = 0; v < 8; ++v) {
            int K = 16 * hb + 2 * v;
            bwb[t].u[v] = pack_bf16(Wb[K * 32 + N], Wb[(K + 1) * 32 + N]);
        }
    }

    for (int it = 0; it < 4; ++it) {
        const int tile = waveGid + it * (NBLK * WAVES);
        const int P0 = tile * 16;
        const int base = (P0 >= 65536) ? 65536 : 0;

        for (int q = 0; q < 16; ++q) {         // x tile -> bf16 pairs
            int pi = lane + 32 * q;
            const float* s = x + (size_t)P0 * 64 + 2 * pi;
            sx[pi] = pack_bf16(s[0], s[1]);
        }
        for (int q = 0; q < 8; ++q) {          // neighbor ids, rel positions, prefetch
            int e = lane + 32 * q;
            int pt = e >> 4;
            int g = base + nidx[(size_t)P0 * 16 + e];
            sidx[e] = g;
            __builtin_prefetch(h_bf + (size_t)g * 32, 0, 3);
#pragma unroll
            for (int d = 0; d < 3; ++d)
                srel[e * 3 + d] = pos[(size_t)(P0 + pt) * 3 + d] - pos[(size_t)g * 3 + d];
        }

        // ---- residual: acc = x @ W_sc + (b_sc + b_out) ----
        FragBF ax0, ax1;
#pragma unroll
        for (int v = 0; v < 8; ++v) {
            int K0 = a_kpair(v, hb);
            ax0.u[v] = sx[(lm * 64 + K0) >> 1];
            ax1.u[v] = sx[(lm * 64 + K0 + 32) >> 1];
        }
        FragF acc[8];
#pragma unroll
        for (int t = 0; t < 8; ++t) {
            int c = t * 16 + lm;
            float bv = sBias[64 + c] + sBias[192 + c];
#pragma unroll
            for (int j = 0; j < 8; ++j) acc[t].f[j] = bv;
            FragBF b0, b1;
#pragma unroll
            for (int v = 0; v < 8; ++v) {
                int K = 16 * hb + 2 * v;
                b0.u[v] = sWsc[c * 32 + (K >> 1)];
                b1.u[v] = sWsc[c * 32 + ((K + 32) >> 1)];
            }
            acc[t].v = wmma_bf16(ax0.v, b0.v, acc[t].v);
            acc[t].v = wmma_bf16(ax1.v, b1.v, acc[t].v);
        }

        // ---- per-point edge MLP + weighted gather-sum ----
        for (int p = 0; p < 16; ++p) {
            // Stage the 16 gathered h rows (64B each) for this point into LDS with
            // full-row B128 transfers: 2 per lane instead of 16 scalar u16 gathers.
            {
#if HAS_ASYNC_LDS
                wait_dscnt0();                 // WAR: prior LDS reads of shg must retire
#pragma unroll
                for (int q = 0; q < 2; ++q) {
                    int ch = lane + 32 * q;    // 64 chunks of 16B
                    int k = ch >> 2, part = ch & 3;
                    const u16t* gp = h_bf + (size_t)sidx[p * 16 + k] * 32 + part * 8;
                    async_ld_lds_b128(gp, &shg[k * 32 + part * 8]);
                }
                wait_asynccnt0();
#else
#pragma unroll
                for (int q = 0; q < 2; ++q) {
                    int ch = lane + 32 * q;
                    int k = ch >> 2, part = ch & 3;
                    const u16t* gp = h_bf + (size_t)sidx[p * 16 + k] * 32 + part * 8;
                    *(uint4*)&shg[k * 32 + part * 8] = *(const uint4*)gp;
                }
#endif
            }

            FragF tacc[2];
#if HAS_WMMA_F32F32
            int e = p * 16 + lm;
            v2f arel;
            arel[0] = hb ? srel[e * 3 + 2] : srel[e * 3 + 0];
            arel[1] = hb ? 0.0f            : srel[e * 3 + 1];
#pragma unroll
            for (int t = 0; t < 2; ++t) {
#pragma unroll
                for (int j = 0; j < 8; ++j) tacc[t].f[j] = 0.0f;
                tacc[t].v = wmma_f32(arel, bwa[t], tacc[t].v);
            }
#else
#pragma unroll
            for (int t = 0; t < 2; ++t)
#pragma unroll
                for (int j = 0; j < 8; ++j) {
                    int ed = p * 16 + j + 8 * hb;
                    tacc[t].f[j] = srel[ed * 3] * wac[t][0] + srel[ed * 3 + 1] * wac[t][1] +
                                   srel[ed * 3 + 2] * wac[t][2];
                }
#endif
            // bias + leaky(0.1); t tile to LDS as bf16 (C-layout -> row-major)
#pragma unroll
            for (int t = 0; t < 2; ++t) {
                int c = t * 16 + lm;
                float bav = sBias[c];
#pragma unroll
                for (int j = 0; j < 8; ++j) {
                    int M = j + 8 * hb;
                    st[M * 32 + c] = f_to_bf16u(lrelu(tacc[t].f[j] + bav, 0.1f));
                }
            }
            FragBF tf;                          // t A-fragment (bf16, K=32)
#pragma unroll
            for (int v = 0; v < 8; ++v) {
                int K0 = a_kpair(v, hb);
                tf.u[v] = *(const u32t*)&st[lm * 32 + K0];
            }
            FragF w0, w1;                       // w = t @ Wb + bb
            {
                float b0v = sBias[32 + lm], b1v = sBias[48 + lm];
#pragma unroll
                for (int j = 0; j < 8; ++j) { w0.f[j] = b0v; w1.f[j] = b1v; }
            }
            w0.v = wmma_bf16(tf.v, bwb[0].v, w0.v);
            w1.v = wmma_bf16(tf.v, bwb[1].v, w1.v);
            // conv[p,c] = sum_k w[k,c] * hg[k,c]  (C-layout dot against LDS-staged rows)
            float vs0 = 0.0f, vs1 = 0.0f;
#pragma unroll
            for (int j = 0; j < 8; ++j) {
                int k = j + 8 * hb;
                vs0 += w0.f[j] * bf16u_to_f(shg[k * 32 + lm]);
                vs1 += w1.f[j] * bf16u_to_f(shg[k * 32 + 16 + lm]);
            }
            vs0 += __shfl_xor(vs0, 16, 32);
            vs1 += __shfl_xor(vs1, 16, 32);
            sconv[p * 32 + lane] = f_to_bf16u(hb ? vs1 : vs0);
        }

        // ---- out: acc += conv @ W_out ; epilogue leaky(0.01) ----
        FragBF ac;
#pragma unroll
        for (int v = 0; v < 8; ++v) {
            int K0 = a_kpair(v, hb);
            ac.u[v] = *(const u32t*)&sconv[lm * 32 + K0];
        }
#pragma unroll
        for (int t = 0; t < 8; ++t) {
            int c = t * 16 + lm;
            FragBF b;
#pragma unroll
            for (int v = 0; v < 8; ++v) {
                int K = 16 * hb + 2 * v;
                b.u[v] = sWout[c * 16 + (K >> 1)];
            }
            acc[t].v = wmma_bf16(ac.v, b.v, acc[t].v);
#pragma unroll
            for (int j = 0; j < 8; ++j) {
                int M = j + 8 * hb;
                out[(size_t)(P0 + M) * 128 + c] = lrelu(acc[t].f[j], 0.01f);
            }
        }
    }
}

// ---------------- host launch ----------------
extern "C" void kernel_launch(void* const* d_in, const int* in_sizes, int n_in,
                              void* d_out, int out_size, void* d_ws, size_t ws_size,
                              hipStream_t stream) {
    (void)in_sizes; (void)n_in; (void)out_size; (void)ws_size;
    const float* x    = (const float*)d_in[0];
    const float* pos  = (const float*)d_in[1];
    const int*   nidx = (const int*)d_in[2];
    const float* Win  = (const float*)d_in[3];
    const float* bin  = (const float*)d_in[4];
    const float* Wa   = (const float*)d_in[5];
    const float* ba   = (const float*)d_in[6];
    const float* Wb   = (const float*)d_in[7];
    const float* bb   = (const float*)d_in[8];
    const float* Wout = (const float*)d_in[9];
    const float* bout = (const float*)d_in[10];
    const float* Wsc  = (const float*)d_in[11];
    const float* bsc  = (const float*)d_in[12];
    u16t* h_bf = (u16t*)d_ws;   // 131072*32 bf16 = 8MB staging for hidden activations

    k_hidden<<<NBLK, 128, 0, stream>>>(x, Win, bin, h_bf);
    k_main<<<NBLK, 128, 0, stream>>>(x, pos, nidx, Wa, ba, Wb, bb, Wout, bout, Wsc, bsc,
                                     h_bf, (float*)d_out);
}
